// LongTermMemory_56684978372930
// MI455X (gfx1250) — matrix operand order repaired and checked by reading, not compiled, not run on hardware
//
#include <hip/hip_runtime.h>

// ---------------- problem constants ----------------
#define BB     4
#define SS     8192
#define DD     2048
#define MM     256
#define NROWS  (BB * SS)          // 32768
#define CHUNKS 64
#define CLEN   128                // CHUNKS*CLEN == SS
#define DECAY  0.99f

// ---------------- GEMM tiling ----------------
#define TILE_ROWS 128             // block rows
#define TILE_COLS 256             // block cols
#define KSTEP     32
#define A_LD      40              // 32 + 8 pad (bf16 units); row = 80 B (16B-aligned)
#define B_LD      40

// CDNA5 async global->LDS copy path (zero VGPR staging), if toolchain exposes it
#if defined(__has_builtin)
#  if __has_builtin(__builtin_amdgcn_global_load_async_to_lds_b128)
#    define ASYNC_B 1
#  endif
#endif
#ifndef ASYNC_B
#  define ASYNC_B 0
#endif

typedef __bf16 bf16;
typedef __attribute__((ext_vector_type(4)))  int    v4i;
typedef __attribute__((ext_vector_type(4)))  float  v4f;
typedef __attribute__((ext_vector_type(8)))  float  v8f;
typedef __attribute__((ext_vector_type(8)))  __bf16 v8bf;
typedef __attribute__((ext_vector_type(16))) __bf16 v16bf;

#if ASYNC_B
typedef __attribute__((address_space(1))) v4i* gv4i_p;   // global v4i*
typedef __attribute__((address_space(3))) v4i* lv4i_p;   // LDS v4i*
#endif

union Frag16 { v16bf v; v8bf h[2]; };

// ---------------- f32 -> bf16 weight convert ----------------
__global__ __launch_bounds__(256) void cvt_f32_bf16(const float* __restrict__ in,
                                                    bf16* __restrict__ out, int n) {
    int i = blockIdx.x * 256 + threadIdx.x;
    if (i < n) out[i] = (bf16)in[i];
}

// ---------------- direct staging (short register live ranges) ----------------
__device__ __forceinline__ void stage_A_direct(const float* __restrict__ A, long row_base,
                                               int lda, int k, int tid, bf16* __restrict__ As) {
    const int r  = tid >> 3;          // 0..31
    const int cq = (tid & 7) * 4;     // 0,4,...,28
#pragma unroll
    for (int it = 0; it < 4; ++it) {
        v4f f = *(const v4f*)(A + (row_base + r + it * 32) * (long)lda + k + cq);
        union { bf16 b[4]; unsigned long long u; } pk;
        pk.b[0] = (bf16)f.x; pk.b[1] = (bf16)f.y;
        pk.b[2] = (bf16)f.z; pk.b[3] = (bf16)f.w;
        *(unsigned long long*)&As[(r + it * 32) * A_LD + cq] = pk.u;   // ds_store_b64
    }
}
__device__ __forceinline__ void stage_B_direct(const bf16* __restrict__ Bw, long col_base,
                                               int ldb, int k, int tid, bf16* __restrict__ Bs) {
    const uint4* sg = (const uint4*)(Bw + (col_base + tid) * (long)ldb + k);
    uint4* dl = (uint4*)&Bs[tid * B_LD];
#pragma unroll
    for (int it = 0; it < 4; ++it) dl[it] = sg[it];                    // ds_store_b128
}

// ---------------- bf16 WMMA GEMM: C[rows,ldc] = A[rows,K](f32) * Bw[cols,K](bf16)^T ----
// grid.x : row block (TILE_ROWS rows), grid.y : col block (TILE_COLS cols)
// 8 waves arranged 2 (row) x 4 (col); wave tile = 64 rows x 64 cols = 4x4 WMMA tiles
__global__ __launch_bounds__(256)
void gemm_bf16(const float* __restrict__ A, int lda,
               const bf16*  __restrict__ Bw, int ldb,
               float* __restrict__ Cmat, int ldc, int K)
{
    __shared__ bf16 As[2][TILE_ROWS * A_LD];
    __shared__ bf16 Bs[2][TILE_COLS * B_LD];

    const int tid   = threadIdx.x;
    const int wave  = tid >> 5;        // 0..7
    const int lane  = tid & 31;
    const int lhalf = lane >> 4;       // 0 or 1
    const int l16   = lane & 15;

    const long row_base = (long)blockIdx.x * TILE_ROWS;
    const long col_base = (long)blockIdx.y * TILE_COLS;
    const int  wave_r   = (wave >> 2) * 64;   // 0 or 64
    const int  wave_c   = (wave & 3) * 64;    // 0,64,128,192

    v8f acc[16];
#pragma unroll
    for (int n = 0; n < 16; ++n)
#pragma unroll
        for (int j = 0; j < 8; ++j) acc[n][j] = 0.0f;

    // prologue: stage tile 0
    stage_A_direct(A, row_base, lda, 0, tid, &As[0][0]);
    stage_B_direct(Bw, col_base, ldb, 0, tid, &Bs[0][0]);
    __syncthreads();

    const int nk = K / KSTEP;
    for (int ik = 0; ik < nk; ++ik) {
        const int  cur   = ik & 1;
        const bool more  = (ik + 1) < nk;
        const int  knext = (ik + 1) * KSTEP;

        // 1) kick off next tile's data movement with zero register cost
        if (more) {
            const float* ap = A + (row_base + (tid >> 3)) * (long)lda + knext + (tid & 7) * 4;
#pragma unroll
            for (int it = 0; it < 4; ++it)
                __builtin_prefetch(ap + (long)it * 32 * lda, 0, 0);    // global_prefetch_b8
#if ASYNC_B
            gv4i_p gp = (gv4i_p)(v4i*)(void*)const_cast<bf16*>(
                            Bw + (col_base + tid) * (long)ldb + knext);
            lv4i_p lp = (lv4i_p)(v4i*)(void*)&Bs[cur ^ 1][tid * B_LD];
            // 64 B/thread global->LDS DMA, ASYNCcnt-tracked, zero VGPR data regs
            __builtin_amdgcn_global_load_async_to_lds_b128(gp, lp,  0, 0);
            __builtin_amdgcn_global_load_async_to_lds_b128(gp, lp, 16, 0);
            __builtin_amdgcn_global_load_async_to_lds_b128(gp, lp, 32, 0);
            __builtin_amdgcn_global_load_async_to_lds_b128(gp, lp, 48, 0);
#else
            __builtin_prefetch(Bw + (col_base + tid) * (long)ldb + knext, 0, 0);
#endif
        }

        // 2) preload B fragments (32x16: lane = col, lanes<16 hold K{0..7,16..23}, lanes>=16 K{8..15,24..31})
        Frag16 bfm[4];
#pragma unroll
        for (int bi = 0; bi < 4; ++bi) {
            const bf16* br = &Bs[cur][(wave_c + bi * 16 + l16) * B_LD + lhalf * 8];
            bfm[bi].h[0] = *(const v8bf*)br;
            bfm[bi].h[1] = *(const v8bf*)(br + 16);
        }

        // 3) 16 WMMAs in 4 bursts; A frag (16x32) loaded per row group
        //    lanes<16 hold K{0..7,16..23}, lanes>=16 K{8..15,24..31}
#pragma unroll
        for (int ai = 0; ai < 4; ++ai) {
            Frag16 af;
            const bf16* ar = &As[cur][(wave_r + ai * 16 + l16) * A_LD + lhalf * 8];
            af.h[0] = *(const v8bf*)ar;          // ds_read_b128
            af.h[1] = *(const v8bf*)(ar + 16);   // ds_read_b128
#pragma unroll
            for (int bi = 0; bi < 4; ++bi)
                acc[ai * 4 + bi] = __builtin_amdgcn_wmma_f32_16x16x32_bf16(
                    false, af.v, false, bfm[bi].v, (short)0,
                    acc[ai * 4 + bi], false, false);
        }

        // 4) stage next tile (short live ranges; loads come from L2 thanks to prefetch)
        if (more) {
            stage_A_direct(A, row_base, lda, knext, tid, &As[cur ^ 1][0]);
#if !ASYNC_B
            stage_B_direct(Bw, col_base, ldb, knext, tid, &Bs[cur ^ 1][0]);
#endif
        }

#if ASYNC_B
        if (more) {
#  if __has_builtin(__builtin_amdgcn_s_wait_asynccnt)
            __builtin_amdgcn_s_wait_asynccnt(0);
#  else
            asm volatile("s_wait_asynccnt 0" ::: "memory");
#  endif
        }
#endif
        __syncthreads();
    }

    // ---- store: VGPR v, lanes<16 -> row v, lanes>=16 -> row v+8; col = tile + lane%16
    const long gr0 = row_base + wave_r + lhalf * 8;
#pragma unroll
    for (int ai = 0; ai < 4; ++ai) {
#pragma unroll
        for (int bi = 0; bi < 4; ++bi) {
            const long col = col_base + wave_c + bi * 16 + l16;
#pragma unroll
            for (int v = 0; v < 8; ++v)
                Cmat[(gr0 + ai * 16 + v) * (long)ldc + col] = acc[ai * 4 + bi][v];
        }
    }
}

// ---------------- scan pass A: per-chunk local EMA scan (zero init), in place ----------
// grid: B*CHUNKS blocks, 256 threads (one per m channel)
__global__ __launch_bounds__(256)
void scan_local(float* __restrict__ xc, float* __restrict__ chunk_end)
{
    const int b = blockIdx.x / CHUNKS;
    const int c = blockIdx.x % CHUNKS;
    const int m = threadIdx.x;
    float mem = 0.0f;
    size_t base = ((size_t)b * SS + (size_t)c * CLEN) * MM + m;
#pragma unroll 4
    for (int i = 0; i < CLEN; ++i) {
        float v = xc[base + (size_t)i * MM];
        mem = DECAY * mem + (1.0f - DECAY) * v;
        xc[base + (size_t)i * MM] = mem;
    }
    chunk_end[((size_t)b * CHUNKS + c) * MM + m] = mem;
}

// ---------------- scan pass B: sequential cross-chunk prefix -------------------------
// grid: B blocks, 256 threads
__global__ __launch_bounds__(256)
void scan_prefix(const float* __restrict__ chunk_end, const float* __restrict__ mem0,
                 float* __restrict__ prefix)
{
    const int b = blockIdx.x;
    const int m = threadIdx.x;
    const float aL = powf(DECAY, (float)CLEN);
    float p = mem0[m];                     // memory_state[0,0,m]
    for (int c = 0; c < CHUNKS; ++c) {
        size_t idx = ((size_t)b * CHUNKS + c) * MM + m;
        prefix[idx] = p;                   // state entering chunk c
        p = aL * p + chunk_end[idx];
    }
}

// ---------------- scan pass C: fixup m_t = local_t + a^{i+1} * p_c -------------------
__global__ __launch_bounds__(256)
void scan_fixup(float* __restrict__ xc, const float* __restrict__ prefix)
{
    size_t idx = (size_t)blockIdx.x * 256 + threadIdx.x;
    const int m = (int)(idx & (MM - 1));
    const size_t t = (idx >> 8) & (SS - 1);
    const int b = (int)(idx >> 21);        // idx / (SS*MM)
    const int c = (int)(t / CLEN);
    const int i = (int)(t % CLEN);
    float p = prefix[((size_t)b * CHUNKS + c) * MM + m];
    xc[idx] += powf(DECAY, (float)(i + 1)) * p;
}

// ---------------- host launch --------------------------------------------------------
extern "C" void kernel_launch(void* const* d_in, const int* in_sizes, int n_in,
                              void* d_out, int out_size, void* d_ws, size_t ws_size,
                              hipStream_t stream)
{
    const float* x      = (const float*)d_in[0];
    const float* w_down = (const float*)d_in[1];
    const float* w_up   = (const float*)d_in[2];
    const float* mem0   = (const float*)d_in[3];
    float*       out    = (float*)d_out;

    char* ws = (char*)d_ws;
    float* xc        = (float*)ws;                        // [B,S,M] f32 = 32 MB
    size_t off = (size_t)NROWS * MM * sizeof(float);
    float* chunk_end = (float*)(ws + off); off += (size_t)BB * CHUNKS * MM * sizeof(float);
    float* prefix    = (float*)(ws + off); off += (size_t)BB * CHUNKS * MM * sizeof(float);
    bf16*  wdb       = (bf16*)(ws + off);  off += (size_t)MM * DD * sizeof(bf16);
    bf16*  wub       = (bf16*)(ws + off);  off += (size_t)DD * MM * sizeof(bf16);

    // 1) weights -> bf16 (L2-resident, ~1 MB each)
    cvt_f32_bf16<<<(MM * DD + 255) / 256, 256, 0, stream>>>(w_down, wdb, MM * DD);
    cvt_f32_bf16<<<(DD * MM + 255) / 256, 256, 0, stream>>>(w_up,   wub, DD * MM);

    // 2) down projection: xc[32768,256] = x[32768,2048] @ w_down^T
    gemm_bf16<<<dim3(NROWS / TILE_ROWS, 1), 256, 0, stream>>>(x, DD, wdb, DD, xc, MM, DD);

    // 3) chunked EMA scan over S
    scan_local <<<BB * CHUNKS, 256, 0, stream>>>(xc, chunk_end);
    scan_prefix<<<BB,          256, 0, stream>>>(chunk_end, mem0, prefix);
    scan_fixup <<<(NROWS * MM) / 256, 256, 0, stream>>>(xc, prefix);

    // 4) up projection: out[32768,2048] = m_c[32768,256] @ w_up^T
    gemm_bf16<<<dim3(NROWS / TILE_ROWS, DD / TILE_COLS), 256, 0, stream>>>(
        xc, MM, wub, MM, out, DD, MM);
}
